// QuantizationLayer_35562329211518
// MI455X (gfx1250) — matrix operand order repaired and checked by reading, not compiled, not run on hardware
//
#include <hip/hip_runtime.h>

typedef __attribute__((ext_vector_type(16))) __bf16 v16bf;
typedef __attribute__((ext_vector_type(8)))  float  v8f;
typedef __attribute__((ext_vector_type(4)))  int    v4i;

#define B_   16
#define C_   256
#define T_   2048
#define K_   1024
#define NTILES 64   // K_/16 codebook tiles
#define KCH    8    // C_/32 k-chunks

#if defined(__has_builtin)
#if __has_builtin(__builtin_amdgcn_global_load_async_to_lds_b128) && \
    __has_builtin(__builtin_amdgcn_s_wait_asynccnt)
#define USE_ASYNC_LDS 1
#endif
#endif
#ifndef USE_ASYNC_LDS
#define USE_ASYNC_LDS 0
#endif

typedef __attribute__((address_space(1))) v4i* as1_v4i;
typedef __attribute__((address_space(3))) v4i* as3_v4i;

__device__ __forceinline__ unsigned short f32_bf16_rne(float f) {
    unsigned u = __float_as_uint(f);
    unsigned r = u + 0x7FFFu + ((u >> 16) & 1u);
    return (unsigned short)(r >> 16);
}
__device__ __forceinline__ float bf16_f32(unsigned short h) {
    return __uint_as_float(((unsigned)h) << 16);
}

union FragBF { v16bf v; unsigned short s[16]; uint4 q[2]; };

// ---- async global->LDS staging helpers (CDNA5 GLOBAL_LOAD_ASYNC_TO_LDS) ----
__device__ __forceinline__ void stage16(const uint4* g, uint4* l) {
#if USE_ASYNC_LDS
    __builtin_amdgcn_global_load_async_to_lds_b128(
        (as1_v4i)(uint4*)g, (as3_v4i)l, 0, 0);
#else
    *l = *g;  // fallback: reg round-trip copy
#endif
}
__device__ __forceinline__ void stage_wait() {
#if USE_ASYNC_LDS
    __builtin_amdgcn_s_wait_asynccnt(0);
#endif
}
__device__ __forceinline__ void stage_tile(const uint4* __restrict__ srcHi,
                                           const uint4* __restrict__ srcLo,
                                           uint4* __restrict__ dst, int tid) {
    stage16(srcHi + tid * 2,     dst + tid * 2);
    stage16(srcHi + tid * 2 + 1, dst + tid * 2 + 1);
    stage16(srcLo + tid * 2,     dst + 512 + tid * 2);
    stage16(srcLo + tid * 2 + 1, dst + 512 + tid * 2 + 1);
}

// ---------------- codebook row norms ----------------
__global__ void vq_cnorm(const float* __restrict__ cb, float* __restrict__ cn) {
    int k = blockIdx.x * blockDim.x + threadIdx.x;
    if (k >= K_) return;
    const float* row = cb + (size_t)k * C_;
    float s = 0.f;
    #pragma unroll 8
    for (int c = 0; c < C_; ++c) s = fmaf(row[c], row[c], s);
    cn[k] = s;
}

// ------- pack codebook into WMMA B-fragment order (hi/lo bf16 split) -------
// B k order chosen here (contiguous; A matches): lane%16 = N column; lanes0-15 K=0..15, lanes16-31
// K=16..31; element e = 2*v+h holds K = base + e.  Tile-contiguous buffers.
__global__ void vq_pack(const float* __restrict__ cb,
                        unsigned short* __restrict__ bhi,
                        unsigned short* __restrict__ blo) {
    int slot = blockIdx.x * blockDim.x + threadIdx.x;     // 0..262143
    int e    =  slot        & 15;
    int lane = (slot >> 4)  & 31;
    int kc   = (slot >> 9)  & 7;
    int tile =  slot >> 12;
    int n = tile * 16 + (lane & 15);
    int k = kc * 32 + ((lane < 16) ? 0 : 16) + e;
    float x = cb[(size_t)n * C_ + k];
    unsigned short hi = f32_bf16_rne(x);
    unsigned short lo = f32_bf16_rne(x - bf16_f32(hi));
    bhi[slot] = hi;
    blo[slot] = lo;
}

// ---------------- main: distances via WMMA + argmin + gather ----------------
__global__ __launch_bounds__(256, 2)
void vq_main(const float* __restrict__ z, const float* __restrict__ cb,
             const float* __restrict__ cnorm,
             const unsigned short* __restrict__ bhi,
             const unsigned short* __restrict__ blo,
             float* __restrict__ zq, float* __restrict__ idx_out) {
    __shared__ unsigned int ldsB[2][4096];   // 2 x (hi 8KB + lo 8KB)
    __shared__ int idxBuf[8][16];

    const int tid  = threadIdx.x;
    const int lane = tid & 31;
    const int wave = tid >> 5;

    const int m0 = blockIdx.x * 128;         // 128 rows per block (same b)
    const int b  = m0 >> 11;                 // T_ = 2048
    const int t0 = m0 & (T_ - 1);
    const int tw = t0 + wave * 16;           // this wave's 16 t-values

    // ---- load this wave's A fragments (16 rows x 256 ch), hi/lo bf16 split.
    // A k order chosen here (contiguous; B matches): lane%16 = M row; lanes0-15 K=0..15, lanes16-31 K=16..31 (K = base + e).
    v16bf ahi[KCH], alo[KCH];
    #pragma unroll
    for (int kc = 0; kc < KCH; ++kc) {
        FragBF H, L;
        #pragma unroll
        for (int e = 0; e < 16; ++e) {
            int v = e >> 1, h = e & 1;
            int k = kc * 32 + ((lane < 16) ? 0 : 16) + e;
            // lanes 0-15 read 16 consecutive t -> 64B coalesced
            float x = z[((size_t)(b * C_ + k) << 11) + tw + (lane & 15)];
            unsigned short hi = f32_bf16_rne(x);
            H.s[e] = hi;
            L.s[e] = f32_bf16_rne(x - bf16_f32(hi));
        }
        ahi[kc] = H.v;
        alo[kc] = L.v;
    }

    float minD[8];
    int   minI[8];
    #pragma unroll
    for (int r = 0; r < 8; ++r) { minD[r] = 3.0e38f; minI[r] = 0; }

    const uint4* baseHi = (const uint4*)bhi;   // 512 uint4 per tile
    const uint4* baseLo = (const uint4*)blo;

    // prologue: stage tile 0 into buffer 0
    stage_tile(baseHi, baseLo, (uint4*)&ldsB[0][0], tid);
    stage_wait();
    __syncthreads();

    for (int t = 0; t < NTILES; ++t) {
        const int buf = t & 1;
        // Kick the next tile's async copy into the other buffer; it overlaps
        // the 24 WMMAs below.  buf^1 was last read in iteration t-1, which all
        // waves completed before the barrier that ended it.
        if (t + 1 < NTILES)
            stage_tile(baseHi + (size_t)(t + 1) * 512,
                       baseLo + (size_t)(t + 1) * 512,
                       (uint4*)&ldsB[buf ^ 1][0], tid);

        float cn = cnorm[t * 16 + (lane & 15)];
        v8f acc = {};
        const uint4* lb = (const uint4*)&ldsB[buf][0];
        #pragma unroll
        for (int kc = 0; kc < KCH; ++kc) {
            FragBF bh, bl;
            int fo = (kc * 32 + lane) * 2;       // uint4 index
            bh.q[0] = lb[fo];       bh.q[1] = lb[fo + 1];
            bl.q[0] = lb[512 + fo]; bl.q[1] = lb[512 + fo + 1];
            acc = __builtin_amdgcn_wmma_f32_16x16x32_bf16(
                false, ahi[kc], false, bh.v, (short)0, acc, false, false);
            acc = __builtin_amdgcn_wmma_f32_16x16x32_bf16(
                false, ahi[kc], false, bl.v, (short)0, acc, false, false);
            acc = __builtin_amdgcn_wmma_f32_16x16x32_bf16(
                false, alo[kc], false, bh.v, (short)0, acc, false, false);
        }

        const int n = t * 16 + (lane & 15);
        #pragma unroll
        for (int r = 0; r < 8; ++r) {
            float s = fmaf(-2.f, acc[r], cn);    // |c|^2 - 2 z.c (|z|^2 const)
            if (s < minD[r]) { minD[r] = s; minI[r] = n; }
        }

        stage_wait();        // this wave's async copies into buf^1 done
        __syncthreads();     // all waves' copies visible + done reading buf
    }

    // ---- argmin across the 16 lanes sharing each accumulator row ----
    #pragma unroll
    for (int m = 1; m < 16; m <<= 1) {
        #pragma unroll
        for (int r = 0; r < 8; ++r) {
            float od = __shfl_xor(minD[r], m, 32);
            int   oi = __shfl_xor(minI[r], m, 32);
            if (od < minD[r] || (od == minD[r] && oi < minI[r])) {
                minD[r] = od; minI[r] = oi;
            }
        }
    }
    if (lane == 0) {
        #pragma unroll
        for (int r = 0; r < 8; ++r) idxBuf[wave][r] = minI[r];       // rows 0-7
    }
    if (lane == 16) {
        #pragma unroll
        for (int r = 0; r < 8; ++r) idxBuf[wave][8 + r] = minI[r];   // rows 8-15
    }
    __syncthreads();

    // ---- write indices (as float, per output dtype) ----
    if (lane < 16)
        idx_out[m0 + wave * 16 + lane] = (float)idxBuf[wave][lane];

    // ---- gather z_q = codebook[idx]; stores 64B-coalesced along t ----
    const int myIdx = idxBuf[wave][lane & 15];
    const int tcol  = tw + (lane & 15);
    const float* crow = cb + (size_t)myIdx * C_;
    #pragma unroll 4
    for (int cc = 0; cc < 128; ++cc) {
        int c = cc * 2 + (lane >> 4);
        zq[((size_t)(b * C_ + c) << 11) + tcol] = crow[c];
    }
}

extern "C" void kernel_launch(void* const* d_in, const int* in_sizes, int n_in,
                              void* d_out, int out_size, void* d_ws, size_t ws_size,
                              hipStream_t stream) {
    const float* z  = (const float*)d_in[0];   // (B, C, T) fp32
    const float* cb = (const float*)d_in[1];   // (K, C)    fp32

    float* zq      = (float*)d_out;                       // B*C*T floats
    float* idx_out = zq + (size_t)B_ * C_ * T_;           // then B*T indices

    char* ws = (char*)d_ws;
    float*          cn  = (float*)ws;                                   // 4KB
    unsigned short* bhi = (unsigned short*)(ws + 4096);                 // 512KB
    unsigned short* blo = (unsigned short*)(ws + 4096 + 524288);        // 512KB

    vq_cnorm<<<4,    256, 0, stream>>>(cb, cn);
    vq_pack <<<1024, 256, 0, stream>>>(cb, bhi, blo);
    vq_main <<<256,  256, 0, stream>>>(z, cb, cn, bhi, blo, zq, idx_out);
}
